// AdaptiveGraphConv_32839319945340
// MI455X (gfx1250) — matrix operand order repaired and checked by reading, not compiled, not run on hardware
//
#include <hip/hip_runtime.h>
#include <cstdint>

#define D    128
#define SWS  144   // LDS stride (halves) for transposed weight tiles (mult of 16 -> 32B aligned rows)
#define SAS  136   // LDS stride (halves) for A tiles (mult of 8 -> 16B aligned chunks)
#define WPB  4     // waves per block in node/gate kernels
#define TPB  (WPB * 32)

typedef __attribute__((ext_vector_type(16))) __bf16 v16bf;
typedef __attribute__((ext_vector_type(8)))  __bf16 v8bf;
typedef __attribute__((ext_vector_type(8)))  float  v8f;

__device__ __forceinline__ float sigmoidf_(float x) { return 1.0f / (1.0f + __expf(-x)); }

__device__ __forceinline__ void zero8(v8f acc[8]) {
#pragma unroll
  for (int t = 0; t < 8; ++t)
#pragma unroll
    for (int i = 0; i < 8; ++i) acc[t][i] = 0.0f;
}

// Stage a 128x128 block of W (rows krows_off..+127 of a row-major [*,128] f32 matrix)
// transposed + converted to bf16 into LDS: sW[n*SWS + k] = W[(krows_off+k)*128 + n].
// Global reads are coalesced along n.
__device__ __forceinline__ void stageWT(__bf16* sW, const float* __restrict__ W, int krows_off) {
  for (int i = threadIdx.x; i < D * D; i += blockDim.x) {
    int k = i >> 7;
    int nc = i & 127;
    sW[nc * SWS + k] = (__bf16)W[(krows_off + k) * D + nc];
  }
}

// acc[t] += A(16x128, LDS bf16, stride SAS) x B (B = W^T rows in LDS, stride SWS), t = 8 N-tiles.
// A frag per lane(g=lane>>4, m=lane&15): halves 0..7 = A[m][k0+16g .. +7], 8..15 = A[m][k0+16g+8 .. +7]
// B frag per lane(g, nn):                halves 0..15 = W[k0+16g .. +15][n0+nn] = sW[n0+nn][k0+16g ..]
__device__ __forceinline__ void gemm_tile(v8f acc[8], const __bf16* sAw, const __bf16* sW, int lane) {
  const int g  = lane >> 4;
  const int nn = lane & 15;
#pragma unroll
  for (int k0 = 0; k0 < D; k0 += 32) {
    v8bf lo = *(const v8bf*)(sAw + nn * SAS + k0 + 16 * g);
    v8bf hi = *(const v8bf*)(sAw + nn * SAS + k0 + 16 * g + 8);
    v16bf av = __builtin_shufflevector(lo, hi, 0,1,2,3,4,5,6,7,8,9,10,11,12,13,14,15);
#pragma unroll
    for (int t = 0; t < 8; ++t) {
      v16bf bv = *(const v16bf*)(sW + (16 * t + nn) * SWS + k0 + 16 * g);
      acc[t] = __builtin_amdgcn_wmma_f32_16x16x32_bf16(false, av, false, bv,
                                                       (short)0, acc[t], false, false);
    }
  }
}

// Row-wise LayerNorm over the C-layout tile: row (r+8g) lives in acc[*][r] across the
// 16 lanes of the half-wave -> partial per lane then 4-step butterfly within the half.
__device__ __forceinline__ void layernorm_tile(v8f acc[8], const float* __restrict__ gamma,
                                               const float* __restrict__ beta,
                                               int lane, bool do_relu) {
  const int nn = lane & 15;
#pragma unroll
  for (int r = 0; r < 8; ++r) {
    float s = 0.f, ss = 0.f;
#pragma unroll
    for (int t = 0; t < 8; ++t) { float v = acc[t][r]; s += v; ss += v * v; }
#pragma unroll
    for (int m = 1; m <= 8; m <<= 1) {
      s  += __shfl_xor(s,  m, 32);
      ss += __shfl_xor(ss, m, 32);
    }
    float mean = s * (1.f / 128.f);
    float var  = ss * (1.f / 128.f) - mean * mean;
    float inv  = rsqrtf(var + 1e-5f);
#pragma unroll
    for (int t = 0; t < 8; ++t) {
      int col = 16 * t + nn;
      float v = (acc[t][r] - mean) * inv * gamma[col] + beta[col];
      if (do_relu) v = fmaxf(v, 0.f);
      acc[t][r] = v;
    }
  }
}

// Kernel 1: per 16-node tile (one wave each):
//   h = relu(LN(x@W1 + b1)); P = h@We1_top + be1; Q = h@We1_bot
__global__ void __launch_bounds__(TPB)
node_kernel(const float* __restrict__ x, const float* __restrict__ W1, const float* __restrict__ b1,
            const float* __restrict__ g1, const float* __restrict__ bt1,
            const float* __restrict__ We1, const float* __restrict__ be1,
            float* __restrict__ hG, float* __restrict__ Pg, float* __restrict__ Qg, int n)
{
  __shared__ __align__(32) __bf16 sW[D * SWS];
  __shared__ __align__(32) __bf16 sA[WPB][16 * SAS];

  const int lane = threadIdx.x & 31;
  const int wv   = threadIdx.x >> 5;
  const int g    = lane >> 4;
  const int nn   = lane & 15;
  const int nTiles = (n + 15) >> 4;
  const int tile = blockIdx.x * WPB + wv;
  const bool valid = tile < nTiles;
  const int rowBase = tile * 16;

  stageWT(sW, W1, 0);
  if (valid) {
    for (int i = lane; i < 16 * D; i += 32) {
      int r = i >> 7, c = i & 127;
      int rr = rowBase + r; if (rr >= n) rr = n - 1;
      sA[wv][r * SAS + c] = (__bf16)x[rr * D + c];
    }
  }
  __syncthreads();

  if (valid) {
    v8f acc[8]; zero8(acc);
    gemm_tile(acc, sA[wv], sW, lane);
#pragma unroll
    for (int t = 0; t < 8; ++t) {
      float bb = b1[16 * t + nn];
#pragma unroll
      for (int r = 0; r < 8; ++r) acc[t][r] += bb;
    }
    layernorm_tile(acc, g1, bt1, lane, true);
    // scatter h (C-layout) to global f32 and to LDS as bf16 A-tile for the next GEMMs
#pragma unroll
    for (int r = 0; r < 8; ++r) {
      int rl = r + 8 * g;
      int rowg = rowBase + rl;
#pragma unroll
      for (int t = 0; t < 8; ++t) {
        int col = 16 * t + nn;
        if (rowg < n) hG[(size_t)rowg * D + col] = acc[t][r];
        sA[wv][rl * SAS + col] = (__bf16)acc[t][r];
      }
    }
  }
  __syncthreads();
  stageWT(sW, We1, 0);            // We1 rows 0..127 (h_src half)
  __syncthreads();
  if (valid) {
    v8f acc[8]; zero8(acc);
    gemm_tile(acc, sA[wv], sW, lane);
#pragma unroll
    for (int t = 0; t < 8; ++t) {
      float bb = be1[16 * t + nn];
#pragma unroll
      for (int r = 0; r < 8; ++r) {
        int rowg = rowBase + r + 8 * g;
        if (rowg < n) Pg[(size_t)rowg * D + 16 * t + nn] = acc[t][r] + bb;
      }
    }
  }
  __syncthreads();
  stageWT(sW, We1, 128);          // We1 rows 128..255 (h_dst half)
  __syncthreads();
  if (valid) {
    v8f acc[8]; zero8(acc);
    gemm_tile(acc, sA[wv], sW, lane);
#pragma unroll
    for (int t = 0; t < 8; ++t)
#pragma unroll
      for (int r = 0; r < 8; ++r) {
        int rowg = rowBase + r + 8 * g;
        if (rowg < n) Qg[(size_t)rowg * D + 16 * t + nn] = acc[t][r];
      }
  }
}

__global__ void zero_kernel(float* __restrict__ p, int count4) {
  float4 z = make_float4(0.f, 0.f, 0.f, 0.f);
  for (int i = blockIdx.x * blockDim.x + threadIdx.x; i < count4; i += gridDim.x * blockDim.x)
    ((float4*)p)[i] = z;
}

// Kernel 2: one wave per edge (grid-stride). Lane covers 4 dims of D=128.
//   ew = sigmoid(relu(P[row]+Q[col]) . We2 + be2);  aggr[col] += ew * h[row]  (L2 f32 atomics)
__global__ void __launch_bounds__(256)
edge_kernel(const long long* __restrict__ ei,
            const float* __restrict__ P, const float* __restrict__ Q,
            const float* __restrict__ hG, const float* __restrict__ We2,
            const float* __restrict__ be2, float* __restrict__ aggr, int E)
{
  const int lane = threadIdx.x & 31;
  const int wave = blockIdx.x * (blockDim.x >> 5) + (threadIdx.x >> 5);
  const int nWaves = gridDim.x * (blockDim.x >> 5);
  const float4 w4 = ((const float4*)We2)[lane];
  const float b2 = be2[0];

  for (int e = wave; e < E; e += nWaves) {
    const int r = (int)ei[e];
    const int c = (int)ei[(size_t)E + e];
    float4 p = ((const float4*)(P + (size_t)r * D))[lane];
    float4 q = ((const float4*)(Q + (size_t)c * D))[lane];
    float t0 = fmaxf(p.x + q.x, 0.f);
    float t1 = fmaxf(p.y + q.y, 0.f);
    float t2 = fmaxf(p.z + q.z, 0.f);
    float t3 = fmaxf(p.w + q.w, 0.f);
    float s = t0 * w4.x + t1 * w4.y + t2 * w4.z + t3 * w4.w;
#pragma unroll
    for (int m = 1; m <= 16; m <<= 1) s += __shfl_xor(s, m, 32);
    float ew = sigmoidf_(s + b2);
    float4 hh = ((const float4*)(hG + (size_t)r * D))[lane];
    float* ap = aggr + (size_t)c * D + 4 * lane;
    __hip_atomic_fetch_add(ap + 0, ew * hh.x, __ATOMIC_RELAXED, __HIP_MEMORY_SCOPE_AGENT);
    __hip_atomic_fetch_add(ap + 1, ew * hh.y, __ATOMIC_RELAXED, __HIP_MEMORY_SCOPE_AGENT);
    __hip_atomic_fetch_add(ap + 2, ew * hh.z, __ATOMIC_RELAXED, __HIP_MEMORY_SCOPE_AGENT);
    __hip_atomic_fetch_add(ap + 3, ew * hh.w, __ATOMIC_RELAXED, __HIP_MEMORY_SCOPE_AGENT);
  }
}

// Kernel 3: gate = sigmoid(h@Wg_top + aggr@Wg_bot + bg); out = LN(gate*aggr + (1-gate)*h)
__global__ void __launch_bounds__(TPB)
gate_kernel(const float* __restrict__ hG, const float* __restrict__ aggr,
            const float* __restrict__ Wg, const float* __restrict__ bg,
            const float* __restrict__ g2, const float* __restrict__ bt2,
            float* __restrict__ out, int n)
{
  __shared__ __align__(32) __bf16 sW[D * SWS];
  __shared__ __align__(32) __bf16 sA[WPB][16 * SAS];

  const int lane = threadIdx.x & 31;
  const int wv   = threadIdx.x >> 5;
  const int g    = lane >> 4;
  const int nn   = lane & 15;
  const int nTiles = (n + 15) >> 4;
  const int tile = blockIdx.x * WPB + wv;
  const bool valid = tile < nTiles;
  const int rowBase = tile * 16;

  stageWT(sW, Wg, 0);
  if (valid) {
    for (int i = lane; i < 16 * D; i += 32) {
      int r = i >> 7, c = i & 127;
      int rr = rowBase + r; if (rr >= n) rr = n - 1;
      sA[wv][r * SAS + c] = (__bf16)hG[(size_t)rr * D + c];
    }
  }
  __syncthreads();

  v8f acc[8]; zero8(acc);
  if (valid) gemm_tile(acc, sA[wv], sW, lane);     // h @ Wg_top
  __syncthreads();
  stageWT(sW, Wg, 128);
  if (valid) {
    for (int i = lane; i < 16 * D; i += 32) {
      int r = i >> 7, c = i & 127;
      int rr = rowBase + r; if (rr >= n) rr = n - 1;
      sA[wv][r * SAS + c] = (__bf16)aggr[(size_t)rr * D + c];
    }
  }
  __syncthreads();
  if (valid) {
    gemm_tile(acc, sA[wv], sW, lane);              // += aggr @ Wg_bot
#pragma unroll
    for (int t = 0; t < 8; ++t) {
      float bb = bg[16 * t + nn];
#pragma unroll
      for (int r = 0; r < 8; ++r) {
        int rowg = rowBase + r + 8 * g;
        int rr = rowg < n ? rowg : n - 1;
        int col = 16 * t + nn;
        float gate = sigmoidf_(acc[t][r] + bb);
        float hv = hG[(size_t)rr * D + col];
        float av = aggr[(size_t)rr * D + col];
        acc[t][r] = gate * av + (1.0f - gate) * hv;
      }
    }
    layernorm_tile(acc, g2, bt2, lane, false);
#pragma unroll
    for (int t = 0; t < 8; ++t)
#pragma unroll
      for (int r = 0; r < 8; ++r) {
        int rowg = rowBase + r + 8 * g;
        if (rowg < n) out[(size_t)rowg * D + 16 * t + nn] = acc[t][r];
      }
  }
}

extern "C" void kernel_launch(void* const* d_in, const int* in_sizes, int n_in,
                              void* d_out, int out_size, void* d_ws, size_t ws_size,
                              hipStream_t stream) {
  const float*     x   = (const float*)d_in[0];
  const long long* ei  = (const long long*)d_in[1];   // edge_index is int64 in the reference
  const float*     W1  = (const float*)d_in[2];
  const float*     b1  = (const float*)d_in[3];
  const float*     g1  = (const float*)d_in[4];
  const float*     bt1 = (const float*)d_in[5];
  const float*     We1 = (const float*)d_in[6];
  const float*     be1 = (const float*)d_in[7];
  const float*     We2 = (const float*)d_in[8];
  const float*     be2 = (const float*)d_in[9];
  // d_in[10..13]: node_attention params — dead code in the reference output, skipped.
  const float*     Wg  = (const float*)d_in[14];
  const float*     bg  = (const float*)d_in[15];
  const float*     g2  = (const float*)d_in[16];
  const float*     bt2 = (const float*)d_in[17];

  const int n = in_sizes[0] / D;
  const int E = in_sizes[1] / 2;

  float* ws   = (float*)d_ws;
  float* hG   = ws;
  float* Pg   = hG + (size_t)n * D;
  float* Qg   = Pg + (size_t)n * D;
  float* aggr = Qg + (size_t)n * D;

  const int nTiles = (n + 15) / 16;
  const int nodeBlocks = (nTiles + WPB - 1) / WPB;

  node_kernel<<<nodeBlocks, TPB, 0, stream>>>(x, W1, b1, g1, bt1, We1, be1, hG, Pg, Qg, n);
  zero_kernel<<<512, 256, 0, stream>>>(aggr, n * D / 4);
  edge_kernel<<<2048, 256, 0, stream>>>(ei, Pg, Qg, hG, We2, be2, aggr, E);
  gate_kernel<<<nodeBlocks, TPB, 0, stream>>>(hG, aggr, Wg, bg, g2, bt2, (float*)d_out, n);
}
